// RGCN_56985626084119
// MI455X (gfx1250) — matrix operand order repaired and checked by reading, not compiled, not run on hardware
//
#include <hip/hip_runtime.h>

// ---------------------------------------------------------------------------
// RGCN forward for gfx1250 (MI455X): bf16 WMMA GEMMs (register-direct
// fragments, fragment-major pre-shuffled weights, pipelined B feed) +
// float-atomic scatter.
// ---------------------------------------------------------------------------

typedef __attribute__((ext_vector_type(16))) __bf16 v16bf;
typedef __attribute__((ext_vector_type(8)))  float  v8f;

enum { FLAG_RELU = 1, FLAG_ACCUM = 2, FLAG_RES = 4 };

#define NCELL 50000
#define NNET  50000

// ---- fp32 weights -> bf16 in WMMA B-fragment-major layout -----------------
// dst slot d: h = d&15, lane = (d>>4)&31, nt = (d>>9)%NT, kb = (d>>9)/NT
// source:     k = kb*32 + 8*(lane>>4) + (h&7) + 16*(h>>3),  n = nt*16 + (lane&15)
__global__ void cvt_weight_frag(const float* __restrict__ W, __bf16* __restrict__ out,
                                int K, int N, int total) {
    int d = blockIdx.x * blockDim.x + threadIdx.x;
    if (d >= total) return;
    int h    = d & 15;
    int lane = (d >> 4) & 31;
    int rest = d >> 9;
    int NT   = N >> 4;
    int nt   = rest % NT;
    int kb   = rest / NT;
    int k = kb * 32 + ((lane >> 4) << 3) + (h & 7) + ((h >> 3) << 4);
    int n = nt * 16 + (lane & 15);
    out[d] = (k < K) ? (__bf16)W[k * N + n] : (__bf16)0.0f;
}

// ---- per-relation degree build --------------------------------------------
__global__ void degree_kernel(const int* __restrict__ src, const int* __restrict__ dst,
                              float* __restrict__ deg_out, float* __restrict__ deg_in, int E) {
    int e = blockIdx.x * blockDim.x + threadIdx.x;
    if (e < E) {
        atomicAdd(&deg_out[src[e]], 1.0f);
        atomicAdd(&deg_in[dst[e]], 1.0f);
    }
}

// ---- edge scatter: agg[dst] += x[src] * deg_out^-0.5 (* ew) ---------------
// 64 threads per edge, 4 contiguous floats each (256 cols).
__global__ void scatter_kernel(const float* __restrict__ x, const int* __restrict__ src,
                               const int* __restrict__ dst, const float* __restrict__ deg_out,
                               const float* __restrict__ ew, float* __restrict__ agg, int E) {
    long long gid = (long long)blockIdx.x * blockDim.x + threadIdx.x;
    int e = (int)(gid >> 6);
    int c = (int)(gid & 63) << 2;
    if (e >= E) return;
    int s = src[e], d = dst[e];
    float scale = rsqrtf(fmaxf(deg_out[s], 1.0f));
    if (ew) scale *= ew[e];
    const float4 v = *(const float4*)(x + (long long)s * 256 + c);
    float* o = agg + (long long)d * 256 + c;
    atomicAdd(o + 0, v.x * scale);
    atomicAdd(o + 1, v.y * scale);
    atomicAdd(o + 2, v.z * scale);
    atomicAdd(o + 3, v.w * scale);
}

// ---- bf16 WMMA GEMM: out[M,N] = epilogue(A[M,K] (*rowscale) @ W[K,N] + b) --
// 256 threads = 8 wave32, 128 rows x N cols per block. No LDS:
//  - A fragment (16x32, rows private to each wave) loaded as 4x float4 per
//    lane directly from global, scaled + cvt to bf16 in registers.
//  - B fragments: contiguous 32B v16bf loads from fragment-major Wfrag,
//    two-stage rotated so one load is always in flight behind the WMMA.
template <int N, int FLAGS>
__global__ __launch_bounds__(256) void gemm_bf16(
    const float* __restrict__ A, const float* __restrict__ deg_in,
    const __bf16* __restrict__ Wfrag, const float* __restrict__ bias,
    const float* __restrict__ residual, float* __restrict__ out,
    int M, int K) {
    constexpr int NT = N / 16;
    const int t    = threadIdx.x;
    const int wave = t >> 5;
    const int lane = t & 31;
    const int lhi  = lane >> 4;           // selects k-half
    const int m    = lane & 15;           // A row within wave tile / C column
    const int row0 = blockIdx.x * 128;
    const int mBase = wave * 16;
    const int grow  = row0 + mBase + m;   // A row this lane streams
    const bool rowOK = grow < M;

    float scale = 1.0f;
    if (deg_in && rowOK) scale = rsqrtf(fmaxf(deg_in[grow], 1.0f));
    const float* Arow = A + (long long)grow * K;

    v8f acc[NT];
#pragma unroll
    for (int i = 0; i < NT; ++i) acc[i] = {};

    const int Kpad = (K + 31) & ~31;
    for (int k0 = 0; k0 < Kpad; k0 += 32) {
        // A fragment: k = 8*lhi + {0..7} and {16..23} relative to k0.
        const int kb = k0 + 8 * lhi;
        float4 a0 = {0.f, 0.f, 0.f, 0.f}, a1 = a0, a2 = a0, a3 = a0;
        if (rowOK) {
            if (kb < K)      a0 = *(const float4*)(Arow + kb);
            if (kb + 4 < K)  a1 = *(const float4*)(Arow + kb + 4);
            if (kb + 16 < K) a2 = *(const float4*)(Arow + kb + 16);
            if (kb + 20 < K) a3 = *(const float4*)(Arow + kb + 20);
        }
        v16bf af;
        af[0]  = (__bf16)(a0.x * scale); af[1]  = (__bf16)(a0.y * scale);
        af[2]  = (__bf16)(a0.z * scale); af[3]  = (__bf16)(a0.w * scale);
        af[4]  = (__bf16)(a1.x * scale); af[5]  = (__bf16)(a1.y * scale);
        af[6]  = (__bf16)(a1.z * scale); af[7]  = (__bf16)(a1.w * scale);
        af[8]  = (__bf16)(a2.x * scale); af[9]  = (__bf16)(a2.y * scale);
        af[10] = (__bf16)(a2.z * scale); af[11] = (__bf16)(a2.w * scale);
        af[12] = (__bf16)(a3.x * scale); af[13] = (__bf16)(a3.y * scale);
        af[14] = (__bf16)(a3.z * scale); af[15] = (__bf16)(a3.w * scale);

        // B fragments, software-pipelined one ahead of the WMMA chain.
        const __bf16* bp = Wfrag + ((size_t)(k0 >> 5) * NT * 512 + (size_t)lane * 16);
        v16bf bcur = *(const v16bf*)bp;
#pragma unroll
        for (int nt = 0; nt < NT; ++nt) {
            v16bf bnext;
            if (nt + 1 < NT) bnext = *(const v16bf*)(bp + (size_t)(nt + 1) * 512);
            acc[nt] = __builtin_amdgcn_wmma_f32_16x16x32_bf16(
                false, af, false, bcur, (short)0, acc[nt], false, false);
            bcur = bnext;
        }
    }

    // Epilogue: C layout is n = lane%16, row = v + 8*(lane>>4).
#pragma unroll
    for (int nt = 0; nt < NT; ++nt) {
#pragma unroll
        for (int v = 0; v < 8; ++v) {
            int mm = v + (lhi << 3);
            int orow = row0 + mBase + mm;
            if (orow < M) {
                int n = nt * 16 + m;
                long long idx = (long long)orow * N + n;
                float val = acc[nt][v] + bias[n];
                if (FLAGS & FLAG_ACCUM) val += out[idx];
                if (FLAGS & FLAG_RES) val += residual[idx];
                if (FLAGS & FLAG_RELU) val = fmaxf(val, 0.0f);
                out[idx] = val;
            }
        }
    }
}

#define GEMM(Nv, Fv, A, din, W, b, res, o, M, K)                                   \
    gemm_bf16<Nv, Fv><<<((M) + 127) / 128, 256, 0, stream>>>(A, din, W, b, res, o, M, K)

// ---------------------------------------------------------------------------
extern "C" void kernel_launch(void* const* d_in, const int* in_sizes, int n_in,
                              void* d_out, int out_size, void* d_ws, size_t ws_size,
                              hipStream_t stream) {
    (void)n_in; (void)out_size; (void)ws_size;
    const float* h_cell = (const float*)d_in[0];
    const float* h_net  = (const float*)d_in[1];
    const int* cc_src = (const int*)d_in[2];
    const int* cc_dst = (const int*)d_in[3];
    const int* cn_src = (const int*)d_in[4];
    const int* cn_dst = (const int*)d_in[5];
    const int* nc_src = (const int*)d_in[6];
    const int* nc_dst = (const int*)d_in[7];
    const int* nn_src = (const int*)d_in[8];
    const int* nn_dst = (const int*)d_in[9];
    const float* nn_ew = (const float*)d_in[10];
    const int E = in_sizes[2];

    // Params (flattened in make_params insertion order, after the 12 inputs).
    const int WI[18] = {12, 14, 16, 18, 20, 22, 24, 26, 28, 30, 32, 34, 36, 38, 40, 42, 44, 46};
    // wb[k]: 0 nlin1 | 1 lin1_c 2 lin1_n | 3 lin11_c 4 lin11_n | 5 lin12_c 6 lin12_n |
    //        7 lin2_c 8 lin2_n | 9 lin21_c 10 lin21_n | 11 lin22_c 12 lin22_n |
    //        13 conv_cc 14 conv_cn 15 conv_nc 16 conv_nn | 17 nlin2
    const int WK[18] = {16, 128, 128, 128, 128, 128, 128, 256, 256,
                        128, 128, 128, 128, 256, 256, 256, 256, 128};
    const int WN[18] = {128, 128, 128, 128, 128, 256, 256, 128, 128,
                        128, 128, 128, 128, 256, 256, 256, 256, 256};
    const float* bias[18];
    for (int i = 0; i < 18; ++i) bias[i] = (const float*)d_in[WI[i] + 1];

    // ---- workspace carve-up (floats) ----
    float* ws = (float*)d_ws;
    size_t off = 0;
    auto alloc = [&](size_t n) { float* p = ws + off; off += n; return p; };
    float* hn0   = alloc((size_t)NNET * 128);
    float* cfeat = alloc((size_t)NCELL * 256);
    float* nfeat = alloc((size_t)NNET * 256);
    float* oc    = alloc((size_t)NCELL * 256);
    float* on    = alloc((size_t)NNET * 256);
    float* agg   = alloc((size_t)50000 * 256);
    float* tmp0  = alloc((size_t)50000 * 128);
    float* tmp1  = alloc((size_t)50000 * 128);
    float* degs  = alloc((size_t)8 * 50000);
    off = (off + 15) & ~(size_t)15;                 // 64B-align weight region
    __bf16* wbase = (__bf16*)(ws + off);

    // ---- shuffle+convert weights into fragment-major bf16 ----
    const __bf16* wb[18];
    {
        size_t woff = 0;
        for (int i = 0; i < 18; ++i) {
            int Kpad = (WK[i] + 31) & ~31;
            int total = Kpad * WN[i];
            __bf16* dst = wbase + woff;
            wb[i] = dst;
            cvt_weight_frag<<<(total + 255) / 256, 256, 0, stream>>>(
                (const float*)d_in[WI[i]], dst, WK[i], WN[i], total);
            woff += (size_t)total;
        }
    }

    // ---- degrees (4 relations, 8 arrays of 50000) ----
    hipMemsetAsync(degs, 0, 8ull * 50000 * sizeof(float), stream);
    const int D = 50000;
    int degGrid = (E + 255) / 256;
    degree_kernel<<<degGrid, 256, 0, stream>>>(cc_src, cc_dst, degs + 0 * D, degs + 1 * D, E);
    degree_kernel<<<degGrid, 256, 0, stream>>>(cn_src, cn_dst, degs + 2 * D, degs + 3 * D, E);
    degree_kernel<<<degGrid, 256, 0, stream>>>(nc_src, nc_dst, degs + 4 * D, degs + 5 * D, E);
    degree_kernel<<<degGrid, 256, 0, stream>>>(nn_src, nn_dst, degs + 6 * D, degs + 7 * D, E);

    // ---- dense front-end ----
    GEMM(128, 0, h_net, nullptr, wb[0], bias[0], nullptr, hn0, NNET, 16);                  // nlin1
    GEMM(128, FLAG_RELU, h_cell, nullptr, wb[1], bias[1], nullptr, tmp0, NCELL, 128);      // lin1_c
    GEMM(128, FLAG_RELU, tmp0, nullptr, wb[3], bias[3], nullptr, tmp1, NCELL, 128);        // lin11_c
    GEMM(256, 0, tmp1, nullptr, wb[5], bias[5], nullptr, cfeat, NCELL, 128);               // lin12_c
    GEMM(128, FLAG_RELU, hn0, nullptr, wb[2], bias[2], nullptr, tmp0, NNET, 128);          // lin1_n
    GEMM(128, FLAG_RELU, tmp0, nullptr, wb[4], bias[4], nullptr, tmp1, NNET, 128);         // lin11_n
    GEMM(256, 0, tmp1, nullptr, wb[6], bias[6], nullptr, nfeat, NNET, 128);                // lin12_n

    // ---- hetero graph conv (4 relations) ----
    int scatGrid = (int)(((long long)E * 64 + 255) / 256);
    size_t aggBytes = (size_t)50000 * 256 * sizeof(float);
    // cc -> oc
    hipMemsetAsync(agg, 0, aggBytes, stream);
    scatter_kernel<<<scatGrid, 256, 0, stream>>>(cfeat, cc_src, cc_dst, degs + 0 * D, nullptr, agg, E);
    GEMM(256, 0, agg, degs + 1 * D, wb[13], bias[13], nullptr, oc, NCELL, 256);
    // nc -> oc (accumulate + relu)
    hipMemsetAsync(agg, 0, aggBytes, stream);
    scatter_kernel<<<scatGrid, 256, 0, stream>>>(nfeat, nc_src, nc_dst, degs + 4 * D, nullptr, agg, E);
    GEMM(256, FLAG_ACCUM | FLAG_RELU, agg, degs + 5 * D, wb[15], bias[15], nullptr, oc, NCELL, 256);
    // cn -> on
    hipMemsetAsync(agg, 0, aggBytes, stream);
    scatter_kernel<<<scatGrid, 256, 0, stream>>>(cfeat, cn_src, cn_dst, degs + 2 * D, nullptr, agg, E);
    GEMM(256, 0, agg, degs + 3 * D, wb[14], bias[14], nullptr, on, NNET, 256);
    // nn -> on (edge weights, accumulate + relu)
    hipMemsetAsync(agg, 0, aggBytes, stream);
    scatter_kernel<<<scatGrid, 256, 0, stream>>>(nfeat, nn_src, nn_dst, degs + 6 * D, nn_ew, agg, E);
    GEMM(256, FLAG_ACCUM | FLAG_RELU, agg, degs + 7 * D, wb[16], bias[16], nullptr, on, NNET, 256);

    // ---- dense back-end ----
    float* out_c = (float*)d_out;                       // [NCELL,128]
    float* out_n = (float*)d_out + (size_t)NCELL * 128; // [NNET,256]
    GEMM(128, FLAG_RELU, oc, nullptr, wb[7], bias[7], nullptr, tmp0, NCELL, 256);          // lin2_c
    GEMM(128, FLAG_RELU, tmp0, nullptr, wb[9], bias[9], nullptr, tmp1, NCELL, 128);        // lin21_c
    GEMM(128, FLAG_RES, tmp1, nullptr, wb[11], bias[11], h_cell, out_c, NCELL, 128);       // lin22_c + residual
    GEMM(128, FLAG_RELU, on, nullptr, wb[8], bias[8], nullptr, tmp0, NNET, 256);           // lin2_n
    GEMM(128, FLAG_RELU, tmp0, nullptr, wb[10], bias[10], nullptr, tmp1, NNET, 128);       // lin21_n
    GEMM(128, FLAG_RES, tmp1, nullptr, wb[12], bias[12], hn0, tmp0, NNET, 128);            // lin22_n + hn0
    GEMM(256, 0, tmp0, nullptr, wb[17], bias[17], nullptr, out_n, NNET, 128);              // nlin2
}
